// MixtureOfExperts_23261542875410
// MI455X (gfx1250) — compile-verified, matrix operand order checked by reading, not run on hardware
//
#include <hip/hip_runtime.h>
#include <hip/hip_bf16.h>

#define T_TOK   8192
#define HID     1024
#define FFN     4096
#define NEXP    8
#define NASSIGN (T_TOK * 2)

#define BM 128
#define BN 128
#define BK 32

typedef unsigned short u16;
typedef __attribute__((ext_vector_type(8)))  __bf16 v8bf;
typedef __attribute__((ext_vector_type(16))) __bf16 v16bf;
typedef __attribute__((ext_vector_type(8)))  float  v8f;

__device__ __forceinline__ u16 f2bf(float f) {
  unsigned u = __float_as_uint(f);
  unsigned r = (u + 0x7FFFu + ((u >> 16) & 1u)) >> 16;   // round-to-nearest-even
  return (u16)r;
}

// 32-bit LDS byte offset (what GLOBAL_LOAD_ASYNC_TO_LDS wants in its VDST VGPR)
__device__ __forceinline__ unsigned ldsOff(const void* p) {
  return (unsigned)(unsigned long long)(__attribute__((address_space(3))) const void*)p;
}

// CDNA5 async memory->LDS copy, 16B per lane, tracked with ASYNCcnt
__device__ __forceinline__ void async_ld_b128(unsigned lds, const u16* g) {
  asm volatile("global_load_async_to_lds_b128 %0, %1, off" :: "v"(lds), "v"(g) : "memory");
}
__device__ __forceinline__ void wait_async0() {
  asm volatile("s_wait_asynccnt 0x0" ::: "memory");
}
__device__ __forceinline__ void wait_ds0() {
  asm volatile("s_wait_dscnt 0x0" ::: "memory");
}

// Build a 16x32 bf16 WMMA A/B fragment: lane holds K=[kb..kb+7] and K=[kb+16..kb+23]
__device__ __forceinline__ v16bf ld_frag(const __bf16* rowp, int kb) {
  v8bf lo = *(const v8bf*)(rowp + kb);
  v8bf hi = *(const v8bf*)(rowp + 16 + kb);
  return __builtin_shufflevector(lo, hi, 0,1,2,3,4,5,6,7,8,9,10,11,12,13,14,15);
}

__global__ void moe_init(float* out, int n, int* counts) {
  int i = blockIdx.x * blockDim.x + threadIdx.x;
  for (; i < n; i += gridDim.x * blockDim.x) out[i] = 0.f;
  if (blockIdx.x == 0 && threadIdx.x < NEXP) counts[threadIdx.x] = 0;
}

__global__ void moe_cvt_bf16(const float* __restrict__ src, u16* __restrict__ dst, int n4) {
  int i = blockIdx.x * blockDim.x + threadIdx.x;
  for (; i < n4; i += gridDim.x * blockDim.x) {
    float4 v = ((const float4*)src)[i];
    ushort4 o;
    o.x = f2bf(v.x); o.y = f2bf(v.y); o.z = f2bf(v.z); o.w = f2bf(v.w);
    ((ushort4*)dst)[i] = o;
  }
}

// One wave per token: 8 dot products of length 1024, top-2 + softmax, bucket scatter.
__global__ __launch_bounds__(256) void moe_router(
    const float* __restrict__ x, const float* __restrict__ rw,
    int* counts, int* btok, float* bwt) {
  __shared__ float srw[NEXP * HID];
  for (int i = threadIdx.x * 4; i < NEXP * HID; i += 256 * 4)
    *(float4*)&srw[i] = *(const float4*)&rw[i];
  __syncthreads();

  const int wave = threadIdx.x >> 5, lane = threadIdx.x & 31;
  const int t = blockIdx.x * 8 + wave;
  const float* xr = x + (size_t)t * HID;

  float s[NEXP];
  #pragma unroll
  for (int e = 0; e < NEXP; ++e) s[e] = 0.f;

  for (int d = lane * 4; d < HID; d += 32 * 4) {
    float4 xv = *(const float4*)(xr + d);
    #pragma unroll
    for (int e = 0; e < NEXP; ++e) {
      float4 wv = *(const float4*)&srw[e * HID + d];
      s[e] += xv.x * wv.x + xv.y * wv.y + xv.z * wv.z + xv.w * wv.w;
    }
  }
  #pragma unroll
  for (int e = 0; e < NEXP; ++e) {
    #pragma unroll
    for (int off = 16; off > 0; off >>= 1) s[e] += __shfl_xor(s[e], off, 32);
  }

  if (lane == 0) {
    int e0 = 0; float v0 = s[0];
    #pragma unroll
    for (int e = 1; e < NEXP; ++e) if (s[e] > v0) { v0 = s[e]; e0 = e; }
    int e1 = (e0 == 0) ? 1 : 0; float v1 = s[e1];
    #pragma unroll
    for (int e = 0; e < NEXP; ++e) if (e != e0 && s[e] > v1) { v1 = s[e]; e1 = e; }
    float z  = expf(v1 - v0);
    float p0 = 1.f / (1.f + z), p1 = z / (1.f + z);
    int pos0 = atomicAdd(&counts[e0], 1);
    btok[e0 * T_TOK + pos0] = t;  bwt[e0 * T_TOK + pos0] = p0;
    int pos1 = atomicAdd(&counts[e1], 1);
    btok[e1 * T_TOK + pos1] = t;  bwt[e1 * T_TOK + pos1] = p1;
  }
}

__global__ void moe_scan(const int* counts, int* offsets) {
  if (threadIdx.x == 0) {
    int a = 0;
    for (int e = 0; e < NEXP; ++e) { offsets[e] = a; a += counts[e]; }
  }
}

__global__ void moe_finalize(const int* __restrict__ counts, const int* __restrict__ offsets,
                             const int* __restrict__ btok, const float* __restrict__ bwt,
                             int* __restrict__ tokArr, float* __restrict__ wgtArr) {
  int tid = blockIdx.x * blockDim.x + threadIdx.x;   // 0 .. NEXP*T_TOK-1
  int e = tid >> 13, j = tid & (T_TOK - 1);
  if (j < counts[e]) {
    int row = offsets[e] + j;
    tokArr[row] = btok[tid];
    wgtArr[row] = bwt[tid];
  }
}

// h[row, f] = gelu( sum_d xb[tok[row], d] * w1b[e, f, d] ), bf16 out
__global__ __launch_bounds__(256) void moe_gemm1(
    const u16* __restrict__ xb, const u16* __restrict__ w1b, u16* __restrict__ hbuf,
    const int* __restrict__ counts, const int* __restrict__ offsets,
    const int* __restrict__ tokArr) {
  const int e   = blockIdx.z;
  const int cnt = counts[e];
  const int m0  = blockIdx.x * BM;
  if (m0 >= cnt) return;
  const int n0  = blockIdx.y * BN;
  const int off = offsets[e];

  __shared__ __align__(16) __bf16 As[2][BM][BK];
  __shared__ __align__(16) __bf16 Bs[2][BN][BK];
  __shared__ int toks[BM];

  const int tid = threadIdx.x;
  if (tid < BM) {
    int j = m0 + tid; if (j >= cnt) j = cnt - 1;
    toks[tid] = tokArr[off + j];
  }
  __syncthreads();

  const int lane = tid & 31;
  const int wv = tid >> 5;
  const int wm = wv & 3, wn = wv >> 2;         // 4 x 2 wave grid, wave tile 32x64
  const int lr = lane & 15, kb = (lane >> 4) * 8;

  // Per-lane async staging coordinates: wave wv fills rows [wv*16, wv*16+16)
  // of both tiles; each lane moves one 16B chunk per instruction.
  const int sr0 = wv * 16 + (lane >> 2);       // rows 0..7 of the wave's slab
  const int sr1 = sr0 + 8;                     // rows 8..15
  const int sc  = (lane & 3) * 8;              // 16B chunk within 64B row

  const u16* gA0 = xb + (size_t)toks[sr0] * HID + sc;
  const u16* gA1 = xb + (size_t)toks[sr1] * HID + sc;
  const u16* gB0 = w1b + (size_t)e * FFN * HID + (size_t)(n0 + sr0) * HID + sc;
  const u16* gB1 = w1b + (size_t)e * FFN * HID + (size_t)(n0 + sr1) * HID + sc;
  const unsigned lA0 = ldsOff(&As[0][sr0][sc]);
  const unsigned lA1 = ldsOff(&As[0][sr1][sc]);
  const unsigned lB0 = ldsOff(&Bs[0][sr0][sc]);
  const unsigned lB1 = ldsOff(&Bs[0][sr1][sc]);
  const unsigned bufStride = (unsigned)(BM * BK * sizeof(__bf16));  // 8 KB

  const v8f zacc = {0.f,0.f,0.f,0.f,0.f,0.f,0.f,0.f};
  v8f acc[2][4];
  #pragma unroll
  for (int mt = 0; mt < 2; ++mt)
    #pragma unroll
    for (int nt = 0; nt < 4; ++nt) acc[mt][nt] = zacc;

  // prologue: fill buffer 0 with tile k=0
  async_ld_b128(lA0, gA0); async_ld_b128(lA1, gA1);
  async_ld_b128(lB0, gB0); async_ld_b128(lB1, gB1);

  int buf = 0;
  for (int k0 = 0; k0 < HID; k0 += BK) {
    wait_async0();                 // my fills of buf done
    __syncthreads();               // everyone's fills of buf done

    v16bf af[2], bfv[4];
    #pragma unroll
    for (int mt = 0; mt < 2; ++mt) af[mt]  = ld_frag(&As[buf][wm * 32 + mt * 16 + lr][0], kb);
    #pragma unroll
    for (int nt = 0; nt < 4; ++nt) bfv[nt] = ld_frag(&Bs[buf][wn * 64 + nt * 16 + lr][0], kb);
    wait_ds0();                    // my reads of buf retired
    __syncthreads();               // everyone's reads retired -> safe to refill other use of buf later

    if (k0 + BK < HID) {           // overlap next tile's fill with the WMMAs below
      unsigned d = (unsigned)(buf ^ 1) * bufStride;
      async_ld_b128(lA0 + d, gA0 + k0 + BK); async_ld_b128(lA1 + d, gA1 + k0 + BK);
      async_ld_b128(lB0 + d, gB0 + k0 + BK); async_ld_b128(lB1 + d, gB1 + k0 + BK);
    }

    #pragma unroll
    for (int mt = 0; mt < 2; ++mt)
      #pragma unroll
      for (int nt = 0; nt < 4; ++nt)
        acc[mt][nt] = __builtin_amdgcn_wmma_f32_16x16x32_bf16(
            false, af[mt], false, bfv[nt], (short)0, acc[mt][nt], false, false);
    buf ^= 1;
  }

  const int rhi = (lane >> 4) * 8;
  #pragma unroll
  for (int mt = 0; mt < 2; ++mt) {
    #pragma unroll
    for (int nt = 0; nt < 4; ++nt) {
      const int col = n0 + wn * 64 + nt * 16 + lr;
      #pragma unroll
      for (int r = 0; r < 8; ++r) {
        int ml = wm * 32 + mt * 16 + rhi + r;
        int j = m0 + ml;
        if (j < cnt) {
          float v = acc[mt][nt][r];
          float g = 0.5f * v * (1.f + erff(v * 0.70710678118f));   // exact gelu
          hbuf[(size_t)(off + j) * FFN + col] = f2bf(g);
        }
      }
    }
  }
}

// out[tok[row], d] += gate[row] * sum_f h[row, f] * w2b[e, d, f]
__global__ __launch_bounds__(256) void moe_gemm2(
    const u16* __restrict__ hbuf, const u16* __restrict__ w2b, float* __restrict__ out,
    const int* __restrict__ counts, const int* __restrict__ offsets,
    const int* __restrict__ tokArr, const float* __restrict__ wgtArr) {
  const int e   = blockIdx.z;
  const int cnt = counts[e];
  const int m0  = blockIdx.x * BM;
  if (m0 >= cnt) return;
  const int n0  = blockIdx.y * BN;
  const int off = offsets[e];

  __shared__ __align__(16) __bf16 As[2][BM][BK];
  __shared__ __align__(16) __bf16 Bs[2][BN][BK];
  __shared__ int   otoks[BM];
  __shared__ float wgts[BM];

  const int tid = threadIdx.x;
  if (tid < BM) {
    int j = m0 + tid; if (j >= cnt) j = cnt - 1;
    otoks[tid] = tokArr[off + j];
    wgts[tid]  = wgtArr[off + j];
  }
  __syncthreads();

  const int lane = tid & 31;
  const int wv = tid >> 5;
  const int wm = wv & 3, wn = wv >> 2;
  const int lr = lane & 15, kb = (lane >> 4) * 8;

  const int sr0 = wv * 16 + (lane >> 2);
  const int sr1 = sr0 + 8;
  const int sc  = (lane & 3) * 8;

  int j0 = m0 + sr0; if (j0 >= cnt) j0 = cnt - 1;
  int j1 = m0 + sr1; if (j1 >= cnt) j1 = cnt - 1;
  const u16* gA0 = hbuf + (size_t)(off + j0) * FFN + sc;
  const u16* gA1 = hbuf + (size_t)(off + j1) * FFN + sc;
  const u16* gB0 = w2b + ((size_t)e * HID + (size_t)(n0 + sr0)) * FFN + sc;
  const u16* gB1 = w2b + ((size_t)e * HID + (size_t)(n0 + sr1)) * FFN + sc;
  const unsigned lA0 = ldsOff(&As[0][sr0][sc]);
  const unsigned lA1 = ldsOff(&As[0][sr1][sc]);
  const unsigned lB0 = ldsOff(&Bs[0][sr0][sc]);
  const unsigned lB1 = ldsOff(&Bs[0][sr1][sc]);
  const unsigned bufStride = (unsigned)(BM * BK * sizeof(__bf16));

  const v8f zacc = {0.f,0.f,0.f,0.f,0.f,0.f,0.f,0.f};
  v8f acc[2][4];
  #pragma unroll
  for (int mt = 0; mt < 2; ++mt)
    #pragma unroll
    for (int nt = 0; nt < 4; ++nt) acc[mt][nt] = zacc;

  async_ld_b128(lA0, gA0); async_ld_b128(lA1, gA1);
  async_ld_b128(lB0, gB0); async_ld_b128(lB1, gB1);

  int buf = 0;
  for (int k0 = 0; k0 < FFN; k0 += BK) {
    wait_async0();
    __syncthreads();

    v16bf af[2], bfv[4];
    #pragma unroll
    for (int mt = 0; mt < 2; ++mt) af[mt]  = ld_frag(&As[buf][wm * 32 + mt * 16 + lr][0], kb);
    #pragma unroll
    for (int nt = 0; nt < 4; ++nt) bfv[nt] = ld_frag(&Bs[buf][wn * 64 + nt * 16 + lr][0], kb);
    wait_ds0();
    __syncthreads();

    if (k0 + BK < FFN) {
      unsigned d = (unsigned)(buf ^ 1) * bufStride;
      async_ld_b128(lA0 + d, gA0 + k0 + BK); async_ld_b128(lA1 + d, gA1 + k0 + BK);
      async_ld_b128(lB0 + d, gB0 + k0 + BK); async_ld_b128(lB1 + d, gB1 + k0 + BK);
    }

    #pragma unroll
    for (int mt = 0; mt < 2; ++mt)
      #pragma unroll
      for (int nt = 0; nt < 4; ++nt)
        acc[mt][nt] = __builtin_amdgcn_wmma_f32_16x16x32_bf16(
            false, af[mt], false, bfv[nt], (short)0, acc[mt][nt], false, false);
    buf ^= 1;
  }

  const int rhi = (lane >> 4) * 8;
  #pragma unroll
  for (int mt = 0; mt < 2; ++mt) {
    #pragma unroll
    for (int nt = 0; nt < 4; ++nt) {
      const int col = n0 + wn * 64 + nt * 16 + lr;
      #pragma unroll
      for (int r = 0; r < 8; ++r) {
        int ml = wm * 32 + mt * 16 + rhi + r;
        int j = m0 + ml;
        if (j < cnt)   // exactly 2 contributions per out element; 2-term fp add is order-invariant
          atomicAdd(&out[(size_t)otoks[ml] * HID + col], acc[mt][nt][r] * wgts[ml]);
      }
    }
  }
}

extern "C" void kernel_launch(void* const* d_in, const int* in_sizes, int n_in,
                              void* d_out, int out_size, void* d_ws, size_t ws_size,
                              hipStream_t stream) {
  (void)in_sizes; (void)n_in; (void)out_size; (void)ws_size;
  const float* x  = (const float*)d_in[0];
  const float* rw = (const float*)d_in[1];
  const float* w1 = (const float*)d_in[2];
  const float* w2 = (const float*)d_in[3];
  float* out = (float*)d_out;

  size_t off = 0;
  auto carve = [&](size_t bytes) -> char* {
    char* p = (char*)d_ws + off;
    off += (bytes + 255) & ~(size_t)255;
    return p;
  };
  u16*   xb   = (u16*)carve((size_t)T_TOK * HID * 2);          //  16 MB
  u16*   w1b  = (u16*)carve((size_t)NEXP * FFN * HID * 2);     //  64 MB (L2-resident)
  u16*   w2b  = (u16*)carve((size_t)NEXP * HID * FFN * 2);     //  64 MB (L2-resident)
  u16*   hbuf = (u16*)carve((size_t)NASSIGN * FFN * 2);        // 128 MB
  int*   btok = (int*)carve((size_t)NEXP * T_TOK * 4);
  float* bwt  = (float*)carve((size_t)NEXP * T_TOK * 4);
  int*   tokA = (int*)carve((size_t)NASSIGN * 4);
  float* wgtA = (float*)carve((size_t)NASSIGN * 4);
  int*   counts  = (int*)carve(64);
  int*   offsets = (int*)carve(64);

  moe_init<<<1024, 256, 0, stream>>>(out, T_TOK * HID, counts);
  moe_cvt_bf16<<<2048, 256, 0, stream>>>(x,  xb,  T_TOK * HID / 4);
  moe_cvt_bf16<<<4096, 256, 0, stream>>>(w1, w1b, NEXP * FFN * HID / 4);
  moe_cvt_bf16<<<4096, 256, 0, stream>>>(w2, w2b, NEXP * HID * FFN / 4);
  moe_router<<<T_TOK / 8, 256, 0, stream>>>(x, rw, counts, btok, bwt);
  moe_scan<<<1, 32, 0, stream>>>(counts, offsets);
  moe_finalize<<<(NEXP * T_TOK) / 256, 256, 0, stream>>>(counts, offsets, btok, bwt, tokA, wgtA);
  moe_gemm1<<<dim3(T_TOK / BM, FFN / BN, NEXP), 256, 0, stream>>>(xb, w1b, hbuf, counts, offsets, tokA);
  moe_gemm2<<<dim3(T_TOK / BM, HID / BN, NEXP), 256, 0, stream>>>(hbuf, w2b, out, counts, offsets, tokA, wgtA);
}
